// MoeExperts_27041114095774
// MI455X (gfx1250) — matrix operand order repaired and checked by reading, not compiled, not run on hardware
//
#include <hip/hip_runtime.h>
#include <hip/hip_bf16.h>

// MoE experts for MI455X (gfx1250):
//   route (ballot compaction) -> bf16 pre-convert -> grouped bf16 WMMA GEMMs
//   with async global->LDS double-buffered tiles -> combine.

typedef __attribute__((ext_vector_type(16))) __bf16 v16bf;
typedef __attribute__((ext_vector_type(8)))  __bf16 v8bf;
typedef __attribute__((ext_vector_type(8)))  float  v8f;
typedef int v4i __attribute__((vector_size(16)));   // matches builtin param type

#define T_ 2048
#define H_ 1024
#define I_ 768
#define E_ 8
#define K_ 2
#define NA (T_ * K_)   // 4096 total (token, expert) assignments

#define TM  64         // rows (assignments) per block tile
#define TN  128        // output columns per block tile
#define TKD 32         // K depth per WMMA step

// ---------------------------------------------------------------------------
// async global->LDS copy (CDNA5 path), with portable fallback
// ---------------------------------------------------------------------------
#if defined(__has_builtin)
#if __has_builtin(__builtin_amdgcn_global_load_async_to_lds_b128)
#define HAS_ASYNC_LDS 1
#endif
#endif

__device__ __forceinline__ void cp_async16(void* lds, const void* g) {
#ifdef HAS_ASYNC_LDS
  __builtin_amdgcn_global_load_async_to_lds_b128(
      (__attribute__((address_space(1))) v4i*)g,
      (__attribute__((address_space(3))) v4i*)lds, 0, 0);
#else
  *(v8bf*)lds = *(const v8bf*)g;
#endif
}

__device__ __forceinline__ void wait_async() {
#ifdef HAS_ASYNC_LDS
#if __has_builtin(__builtin_amdgcn_s_wait_asynccnt)
  __builtin_amdgcn_s_wait_asynccnt(0);
#else
  asm volatile("s_wait_asynccnt 0x0" ::: "memory");
#endif
#endif
}

// ---------------------------------------------------------------------------
// helpers
// ---------------------------------------------------------------------------
__device__ __forceinline__ v16bf join8(v8bf lo, v8bf hi) {
  v16bf r;
#pragma unroll
  for (int i = 0; i < 8; i++) { r[i] = lo[i]; r[i + 8] = hi[i]; }
  return r;
}

__device__ __forceinline__ v8bf cvt8(float4 a, float4 b) {
  v8bf r;
  r[0] = (__bf16)a.x; r[1] = (__bf16)a.y; r[2] = (__bf16)a.z; r[3] = (__bf16)a.w;
  r[4] = (__bf16)b.x; r[5] = (__bf16)b.y; r[6] = (__bf16)b.z; r[7] = (__bf16)b.w;
  return r;
}

// ---------------------------------------------------------------------------
// 0) fp32 -> bf16 streaming convert (one pass; removes cvt from GEMM loops)
// ---------------------------------------------------------------------------
__global__ __launch_bounds__(256)
void cvt_bf16(const float* __restrict__ src, __bf16* __restrict__ dst, int n8) {
  int i = blockIdx.x * blockDim.x + threadIdx.x;
  if (i >= n8) return;
  const float* s = src + (size_t)i * 8;
  *(v8bf*)(dst + (size_t)i * 8) = cvt8(*(const float4*)s, *(const float4*)(s + 4));
}

// ---------------------------------------------------------------------------
// 1) Routing: deterministic ballot-compaction (no atomics).
// ---------------------------------------------------------------------------
__global__ __launch_bounds__(256)
void moe_route(const int* __restrict__ topk_idx,
               int* __restrict__ counts, int* __restrict__ offs,
               int* __restrict__ assign) {
  __shared__ int s_cnt[E_];
  __shared__ int s_off[E_ + 1];
  const int tid  = threadIdx.x;
  const int w    = tid >> 5;   // expert owned by this wave
  const int lane = tid & 31;

  int c = 0;
  for (int base = 0; base < NA; base += 32) {
    int e = topk_idx[base + lane];
    c += __popc((unsigned)__ballot(e == w));
  }
  if (lane == 0) s_cnt[w] = c;
  __syncthreads();
  if (tid == 0) {
    int o = 0;
    for (int i = 0; i < E_; i++) { s_off[i] = o; o += s_cnt[i]; }
    s_off[E_] = o;
  }
  __syncthreads();

  int pos = s_off[w];
  for (int base = 0; base < NA; base += 32) {
    int a = base + lane;
    int e = topk_idx[a];
    unsigned mm = (unsigned)__ballot(e == w);
    if (e == w) {
      unsigned before = mm & ((1u << lane) - 1u);
      assign[pos + __popc(before)] = a;
    }
    pos += __popc(mm);
  }
  if (tid < E_) { counts[tid] = s_cnt[tid]; offs[tid] = s_off[tid]; }
  if (tid == 0) offs[E_] = s_off[E_];
}

// ---------------------------------------------------------------------------
// 2) gate_up GEMM + SiLU fuse (bf16 in, WMMA f32 acc, bf16 act out).
//    Double-buffered async tiles. grid = (I/TN, NA/TM, E), 256 thr = 8 waves.
// ---------------------------------------------------------------------------
__global__ __launch_bounds__(256)
void moe_gateup(const __bf16* __restrict__ hb,
                const __bf16* __restrict__ wgub,
                const int* __restrict__ counts, const int* __restrict__ offs,
                const int* __restrict__ assign,
                __bf16* __restrict__ act) {
  const int e   = blockIdx.z;
  const int cnt = counts[e];
  const int m0  = blockIdx.y * TM;
  if (m0 >= cnt) return;
  const int off = offs[e];
  const int n0  = blockIdx.x * TN;

  __shared__ __align__(64) __bf16 sA [2][TM][TKD];   // 2 x 4 KB
  __shared__ __align__(64) __bf16 sBg[2][TN][TKD];   // 2 x 8 KB
  __shared__ __align__(64) __bf16 sBu[2][TN][TKD];   // 2 x 8 KB
  __shared__ int sTok[TM];

  const int tid = threadIdx.x;
  if (tid < TM) {
    int gr = m0 + tid;
    sTok[tid] = (gr < cnt) ? (assign[off + gr] >> 1) : 0;  // invalid -> row 0 (masked at store)
  }
  __syncthreads();

  const int wid = tid >> 5, lane = tid & 31;
  const int wm = wid >> 2, wn = wid & 3;       // 2x4 wave grid (32x32 per wave)
  const int lr = lane & 15, lh = lane >> 4;
  const int ar = tid >> 2, ac = (tid & 3) * 8; // A fill: 16B per thread
  const int bj = tid >> 1, bc = (tid & 1) * 16;// B fill: 32B per thread

  const __bf16* aSrc = hb + (size_t)sTok[ar] * H_ + ac;
  const __bf16* gSrc = wgub + ((size_t)e * 2 * I_ + (n0 + bj)) * H_ + bc;
  const __bf16* uSrc = gSrc + (size_t)I_ * H_;

  v8f accg[2][2] = {};
  v8f accu[2][2] = {};

  // prologue: issue tile 0
  cp_async16(&sA[0][ar][ac],       aSrc);
  cp_async16(&sBg[0][bj][bc],      gSrc);
  cp_async16(&sBg[0][bj][bc + 8],  gSrc + 8);
  cp_async16(&sBu[0][bj][bc],      uSrc);
  cp_async16(&sBu[0][bj][bc + 8],  uSrc + 8);

  const int NSTEP = H_ / TKD;   // 32
  for (int step = 0; step < NSTEP; ++step) {
    wait_async();          // this wave's tile landed
    __syncthreads();       // everyone's tile landed
    if (step + 1 < NSTEP) {            // overlap next DMA with compute
      const int nb = (step + 1) & 1;
      const int kk = (step + 1) * TKD;
      cp_async16(&sA[nb][ar][ac],       aSrc + kk);
      cp_async16(&sBg[nb][bj][bc],      gSrc + kk);
      cp_async16(&sBg[nb][bj][bc + 8],  gSrc + kk + 8);
      cp_async16(&sBu[nb][bj][bc],      uSrc + kk);
      cp_async16(&sBu[nb][bj][bc + 8],  uSrc + kk + 8);
    }
    const int cb = step & 1;

    v16bf af[2], bgf[2], buf2[2];
#pragma unroll
    for (int ms = 0; ms < 2; ms++) {   // A frag: two 16B chunks per lane
      int row = wm * 32 + ms * 16 + lr;
      af[ms] = join8(*(const v8bf*)&sA[cb][row][lh * 16],
                     *(const v8bf*)&sA[cb][row][lh * 16 + 8]);
    }
#pragma unroll
    for (int ns = 0; ns < 2; ns++) {   // B frag: one contiguous 32B chunk
      int col = wn * 32 + ns * 16 + lr;
      bgf[ns]  = *(const v16bf*)&sBg[cb][col][lh * 16];
      buf2[ns] = *(const v16bf*)&sBu[cb][col][lh * 16];
    }
#pragma unroll
    for (int ms = 0; ms < 2; ms++)
#pragma unroll
      for (int ns = 0; ns < 2; ns++) {
        accg[ms][ns] = __builtin_amdgcn_wmma_f32_16x16x32_bf16(
            false, af[ms], false, bgf[ns], (short)0, accg[ms][ns], false, false);
        accu[ms][ns] = __builtin_amdgcn_wmma_f32_16x16x32_bf16(
            false, af[ms], false, buf2[ns], (short)0, accu[ms][ns], false, false);
      }
  }

  // epilogue: silu(gate) * up -> bf16 act
#pragma unroll
  for (int ms = 0; ms < 2; ms++)
#pragma unroll
    for (int ns = 0; ns < 2; ns++) {
      v8f g = accg[ms][ns], u = accu[ms][ns];
      int col = n0 + wn * 32 + ns * 16 + lr;
#pragma unroll
      for (int r = 0; r < 8; r++) {
        int gr = m0 + wm * 32 + ms * 16 + lh * 8 + r;  // C layout: M = r + 8*lh
        if (gr < cnt) {
          float gv = g[r];
          float a  = (gv / (1.0f + __expf(-gv))) * u[r];
          act[(size_t)(off + gr) * I_ + col] = (__bf16)a;
        }
      }
    }
}

// ---------------------------------------------------------------------------
// 3) down GEMM: y[a, h] = act_row @ Wd[e,h,:]^T. grid = (H/TN, NA/TM, E).
// ---------------------------------------------------------------------------
__global__ __launch_bounds__(256)
void moe_down(const __bf16* __restrict__ act,
              const __bf16* __restrict__ wdb,
              const int* __restrict__ counts, const int* __restrict__ offs,
              const int* __restrict__ assign,
              float* __restrict__ y) {
  const int e   = blockIdx.z;
  const int cnt = counts[e];
  const int m0  = blockIdx.y * TM;
  if (m0 >= cnt) return;
  const int off = offs[e];
  const int n0  = blockIdx.x * TN;

  __shared__ __align__(64) __bf16 sA[2][TM][TKD];
  __shared__ __align__(64) __bf16 sB[2][TN][TKD];
  __shared__ int sAsn[TM];

  const int tid = threadIdx.x;
  if (tid < TM) {
    int gr = m0 + tid;
    sAsn[tid] = (gr < cnt) ? assign[off + gr] : -1;
  }
  __syncthreads();

  const int wid = tid >> 5, lane = tid & 31;
  const int wm = wid >> 2, wn = wid & 3;
  const int lr = lane & 15, lh = lane >> 4;
  const int ar = tid >> 2, ac = (tid & 3) * 8;
  const int bj = tid >> 1, bc = (tid & 1) * 16;

  const int grA = m0 + ar;
  const __bf16* aSrc = act + (size_t)(off + (grA < cnt ? grA : 0)) * I_ + ac;
  const __bf16* bSrc = wdb + ((size_t)e * H_ + (n0 + bj)) * I_ + bc;

  v8f acc[2][2] = {};

  cp_async16(&sA[0][ar][ac],      aSrc);
  cp_async16(&sB[0][bj][bc],      bSrc);
  cp_async16(&sB[0][bj][bc + 8],  bSrc + 8);

  const int NSTEP = I_ / TKD;   // 24
  for (int step = 0; step < NSTEP; ++step) {
    wait_async();
    __syncthreads();
    if (step + 1 < NSTEP) {
      const int nb = (step + 1) & 1;
      const int kk = (step + 1) * TKD;
      cp_async16(&sA[nb][ar][ac],      aSrc + kk);
      cp_async16(&sB[nb][bj][bc],      bSrc + kk);
      cp_async16(&sB[nb][bj][bc + 8],  bSrc + kk + 8);
    }
    const int cb = step & 1;

    v16bf af[2], bf[2];
#pragma unroll
    for (int ms = 0; ms < 2; ms++) {
      int row = wm * 32 + ms * 16 + lr;
      af[ms] = join8(*(const v8bf*)&sA[cb][row][lh * 16],
                     *(const v8bf*)&sA[cb][row][lh * 16 + 8]);
    }
#pragma unroll
    for (int ns = 0; ns < 2; ns++) {
      int col = wn * 32 + ns * 16 + lr;
      bf[ns] = *(const v16bf*)&sB[cb][col][lh * 16];
    }
#pragma unroll
    for (int ms = 0; ms < 2; ms++)
#pragma unroll
      for (int ns = 0; ns < 2; ns++)
        acc[ms][ns] = __builtin_amdgcn_wmma_f32_16x16x32_bf16(
            false, af[ms], false, bf[ns], (short)0, acc[ms][ns], false, false);
  }

#pragma unroll
  for (int ms = 0; ms < 2; ms++)
#pragma unroll
    for (int ns = 0; ns < 2; ns++) {
      int col = n0 + wn * 32 + ns * 16 + lr;
#pragma unroll
      for (int r = 0; r < 8; r++) {
        int rl = wm * 32 + ms * 16 + lh * 8 + r;
        int gr = m0 + rl;
        if (gr < cnt) {
          int a = sAsn[rl];
          y[(size_t)a * H_ + col] = acc[ms][ns][r];
        }
      }
    }
}

// ---------------------------------------------------------------------------
// 4) combine: out[t,h] = w0*y[2t,h] + w1*y[2t+1,h]
// ---------------------------------------------------------------------------
__global__ __launch_bounds__(256)
void moe_combine(const float* __restrict__ y,
                 const float* __restrict__ topw,
                 float* __restrict__ out) {
  int idx = blockIdx.x * blockDim.x + threadIdx.x;
  if (idx >= T_ * H_) return;
  int t = idx >> 10;            // H_ == 1024
  int h = idx & (H_ - 1);
  float w0 = topw[t * K_ + 0];
  float w1 = topw[t * K_ + 1];
  out[idx] = w0 * y[(size_t)(t * K_ + 0) * H_ + h]
           + w1 * y[(size_t)(t * K_ + 1) * H_ + h];
}

// ---------------------------------------------------------------------------
extern "C" void kernel_launch(void* const* d_in, const int* in_sizes, int n_in,
                              void* d_out, int out_size, void* d_ws, size_t ws_size,
                              hipStream_t stream) {
  const float* hidden = (const float*)d_in[0];   // (T, H)
  const float* wgu    = (const float*)d_in[1];   // (E, 2I, H)
  const float* wd     = (const float*)d_in[2];   // (E, H, I)
  const int*   tki    = (const int*)d_in[3];     // (T, K)
  const float* tkw    = (const float*)d_in[4];   // (T, K)
  float* out = (float*)d_out;

  // workspace layout (~65 MB total)
  int* counts   = (int*)d_ws;                                 // 8
  int* offs     = counts + 8;                                 // 9
  int* assign   = counts + 32;                                // 4096
  __bf16* hidb  = (__bf16*)((char*)d_ws + 16512);             // T*H      (4.19 MB)
  __bf16* wgub  = hidb + (size_t)T_ * H_;                     // E*2I*H   (25.2 MB)
  __bf16* wdb   = wgub + (size_t)E_ * 2 * I_ * H_;            // E*H*I    (12.6 MB)
  __bf16* act   = wdb  + (size_t)E_ * H_ * I_;                // NA*I     (6.29 MB)
  float*  y     = (float*)(act + (size_t)NA * I_);            // NA*H f32 (16.8 MB)

  moe_route<<<1, 256, 0, stream>>>(tki, counts, offs, assign);

  // bf16 pre-convert (hidden, gate_up, down)
  {
    int n8h = T_ * H_ / 8;                 // 262144
    int n8g = E_ * 2 * I_ * H_ / 8;        // 1572864
    int n8d = E_ * H_ * I_ / 8;            // 786432
    cvt_bf16<<<(n8h + 255) / 256, 256, 0, stream>>>(hidden, hidb, n8h);
    cvt_bf16<<<(n8g + 255) / 256, 256, 0, stream>>>(wgu, wgub, n8g);
    cvt_bf16<<<(n8d + 255) / 256, 256, 0, stream>>>(wd, wdb, n8d);
  }

  dim3 g1(I_ / TN, NA / TM, E_);      // (6, 64, 8); dead tiles exit on count check
  moe_gateup<<<g1, 256, 0, stream>>>(hidb, wgub, counts, offs, assign, act);

  dim3 g2(H_ / TN, NA / TM, E_);      // (8, 64, 8)
  moe_down<<<g2, 256, 0, stream>>>(act, wdb, counts, offs, assign, y);

  moe_combine<<<(T_ * H_ + 255) / 256, 256, 0, stream>>>(y, tkw, out);
}
